// FragmentGFlowNet_40166534152622
// MI455X (gfx1250) — matrix operand order repaired and checked by reading, not compiled, not run on hardware
//
#include <hip/hip_runtime.h>

// ---------------------------------------------------------------------------
// FragmentGFlowNet forward for MI455X (gfx1250, wave32, WMMA f32<-f16).
// State arrays (102MB f32 + 51MB f16) are L2-resident (192MB L2); per-graph
// edge aggregation is done in LDS (graphs are contiguous: 20 nodes/40 edges).
// All GEMMs use v_wmma_f32_16x16x32_f16, one 16x16 C tile per wave.
// ---------------------------------------------------------------------------

#define DEVFN __device__ __forceinline__

typedef _Float16 half_t;
typedef __attribute__((ext_vector_type(16))) _Float16 v16h;
typedef __attribute__((ext_vector_type(8)))  _Float16 v8h;
typedef __attribute__((ext_vector_type(8)))  float    v8f;

static constexpr int H        = 128;
static constexpr int G        = 10000;
static constexpr int NPG      = 20;
static constexpr int N        = G * NPG;   // 200000
static constexpr int EPG      = 40;
static constexpr int E        = G * EPG;   // 400000
static constexpr int SPG      = 10;
static constexpr int S        = G * SPG;   // 100000
static constexpr int OUT_STEM = 105;
static constexpr int STEPS    = 12;

// ---------------- WMMA helpers -------------------------------------------

DEVFN v8f wmma(v16h a, v16h b, v8f c) {
  return __builtin_amdgcn_wmma_f32_16x16x32_f16(false, a, false, b,
                                                (short)0, c, false, false);
}

DEVFN v16h cat8(v8h lo, v8h hi) {
  return __builtin_shufflevector(lo, hi, 0, 1, 2, 3, 4, 5, 6, 7,
                                 8, 9, 10, 11, 12, 13, 14, 15);
}

// A fragment (16x32 f16): per ISA layout lane holds M=lane&15 and two runs of
// 8 consecutive K: [hf*8, hf*8+8) and [16+hf*8, 16+hf*8+8).
DEVFN v16h load_afrag(const half_t* rows, int ld, int kb, int lane) {
  const half_t* p = rows + (size_t)(lane & 15) * ld + kb + ((lane >> 4) * 8);
  v8h lo = *(const v8h*)(p);
  v8h hi = *(const v8h*)(p + 16);
  return cat8(lo, hi);
}

// B fragment (32x16 f16): lane holds N=lane&15 and two runs of 8 consecutive
// K: [hf*8, hf*8+8) and [16+hf*8, 16+hf*8+8).  W is stored [N][K]
// (output-feature-major) so this is two 16B loads per lane.
DEVFN v16h load_bfrag(const half_t* W, int ldk, int nb, int kb, int lane) {
  const half_t* p = W + (size_t)(nb + (lane & 15)) * ldk + kb + ((lane >> 4) * 8);
  v8h lo = *(const v8h*)(p);
  v8h hi = *(const v8h*)(p + 16);
  return cat8(lo, hi);
}

DEVFN v8f splat8(float x) {
  v8f v;
#pragma unroll
  for (int i = 0; i < 8; ++i) v[i] = x;
  return v;
}

DEVFN float lrelu(float x) { return x > 0.f ? x : 0.01f * x; }
DEVFN float sigmoidf_(float x) { return 1.f / (1.f + __expf(-x)); }
DEVFN float tanhf_(float x) {
  x = fminf(fmaxf(x, -15.f), 15.f);
  float e = __expf(2.f * x);
  return (e - 1.f) / (e + 1.f);
}

// ---------------- small prep kernels -------------------------------------

__global__ void k_zero_f32(float* p, int n) {
  int i = blockIdx.x * blockDim.x + threadIdx.x;
  if (i < n) p[i] = 0.f;
}

__global__ void k_deg(const int* __restrict__ edst, float* __restrict__ deg, int e) {
  int i = blockIdx.x * blockDim.x + threadIdx.x;
  if (i < e) atomicAdd(&deg[edst[i]], 1.0f);
}

__global__ void k_cvt(const float* __restrict__ s, half_t* __restrict__ d, int n) {
  int i = blockIdx.x * blockDim.x + threadIdx.x;
  if (i < n) d[i] = (half_t)s[i];
}

// s is [K][Nn] row-major -> d is [Nn][K] (packed for B-fragment loads)
__global__ void k_cvtT(const float* __restrict__ s, half_t* __restrict__ d,
                       int K, int Nn) {
  int i = blockIdx.x * blockDim.x + threadIdx.x;
  if (i < K * Nn) {
    int k = i / Nn, n = i % Nn;
    d[n * K + k] = (half_t)s[i];
  }
}

// s2p_w3 [128][105] -> [112][128] transposed + zero-padded
__global__ void k_cvt_w3(const float* __restrict__ s, half_t* __restrict__ d) {
  int i = blockIdx.x * blockDim.x + threadIdx.x;
  if (i < 112 * 128) {
    int n = i >> 7, k = i & 127;
    d[i] = (n < OUT_STEM) ? (half_t)s[k * OUT_STEM + n] : (half_t)0.0f;
  }
}

// ---------------- embedding + f2e MLP ------------------------------------

__global__ __launch_bounds__(256) void k_embed_f2e(
    const int* __restrict__ frag_ids, const half_t* __restrict__ fragE,
    const half_t* __restrict__ w1T, const float* __restrict__ b1,
    const half_t* __restrict__ w2T, const float* __restrict__ b2,
    float* __restrict__ h, half_t* __restrict__ hb, int numTiles) {
  __shared__ __align__(16) half_t lds[8][16][H];
  int lane = threadIdx.x & 31, wave = threadIdx.x >> 5;
  int tile = blockIdx.x * 8 + wave;
  if (tile >= numTiles) return;
  int base = tile * 16;
  int n0 = lane & 15, hf = lane >> 4;
  half_t* X = &lds[wave][0][0];

  for (int r = 0; r < 16; ++r) {
    const uint2* src = (const uint2*)(fragE + (size_t)frag_ids[base + r] * H);
    ((uint2*)(X + r * H))[lane] = src[lane];
  }
  v16h a[4];
#pragma unroll
  for (int kc = 0; kc < 4; ++kc) a[kc] = load_afrag(X, H, kc * 32, lane);
  for (int f = 0; f < 8; ++f) {
    v8f c = splat8(b1[f * 16 + n0]);
#pragma unroll
    for (int kc = 0; kc < 4; ++kc)
      c = wmma(a[kc], load_bfrag(w1T, H, f * 16, kc * 32, lane), c);
#pragma unroll
    for (int r = 0; r < 8; ++r)
      X[(r + 8 * hf) * H + f * 16 + n0] = (half_t)lrelu(c[r]);
  }
  v16h a2[4];
#pragma unroll
  for (int kc = 0; kc < 4; ++kc) a2[kc] = load_afrag(X, H, kc * 32, lane);
  for (int f = 0; f < 8; ++f) {
    v8f c = splat8(b2[f * 16 + n0]);
#pragma unroll
    for (int kc = 0; kc < 4; ++kc)
      c = wmma(a2[kc], load_bfrag(w2T, H, f * 16, kc * 32, lane), c);
#pragma unroll
    for (int r = 0; r < 8; ++r) {
      size_t idx = (size_t)(base + r + 8 * hf) * H + f * 16 + n0;
      h[idx] = c[r];
      hb[idx] = (half_t)c[r];
    }
  }
}

// ---------------- per-graph edge aggregation (LDS, no global atomics) ----

__global__ __launch_bounds__(256) void k_edge_agg(
    const int* __restrict__ esrc, const int* __restrict__ edst,
    const int* __restrict__ ebond, const float* __restrict__ bond_emb,
    const float* __restrict__ h, float* __restrict__ agg) {
  __shared__ float lagg[NPG * H];  // 10 KB
  int g = blockIdx.x;
  int t = threadIdx.x;
  for (int i = t; i < NPG * H; i += 256) lagg[i] = 0.f;
  __syncthreads();
  int lane = t & 31, wave = t >> 5;
  int ebase = g * EPG + wave * (EPG / 8);
  for (int i = 0; i < EPG / 8; ++i) {
    int e = ebase + i;
    int src = esrc[e];
    int dl = edst[e] - g * NPG;
    float4 hv = ((const float4*)(h + (size_t)src * H))[lane];
    float4 e0 = ((const float4*)(bond_emb + (size_t)ebond[2 * e] * H))[lane];
    float s = hv.x * e0.x + hv.y * e0.y + hv.z * e0.z + hv.w * e0.w;
#pragma unroll
    for (int off = 1; off < 32; off <<= 1) s += __shfl_xor(s, off, 32);
    float4 e1 = ((const float4*)(bond_emb + (size_t)ebond[2 * e + 1] * H))[lane];
    float* dp = lagg + dl * H + lane * 4;
    atomicAdd(dp + 0, s * e1.x);
    atomicAdd(dp + 1, s * e1.y);
    atomicAdd(dp + 2, s * e1.z);
    atomicAdd(dp + 3, s * e1.w);
  }
  __syncthreads();
  float* ag = agg + (size_t)g * NPG * H;
  for (int i = t; i < NPG * H; i += 256) ag[i] = lagg[i];
}

// ---------------- fused GRU node update ----------------------------------
// m = lrelu(agg*inv_deg + h@w_root + b_root)
// gi = m@wihT+bih ; gh = h@whhT+bhh ; gate math ; h' = (1-z)*n + z*h
// 224 v_wmma per 16-node tile. Note out==h throughout the loop.

__global__ __launch_bounds__(256) void k_gru(
    const float* __restrict__ agg, const float* __restrict__ deg,
    const half_t* __restrict__ wrootT, const float* __restrict__ broot,
    const half_t* __restrict__ wihT, const float* __restrict__ bih,
    const half_t* __restrict__ whhT, const float* __restrict__ bhh,
    float* __restrict__ h, half_t* __restrict__ hb, int numTiles) {
  __shared__ __align__(16) half_t lds[8][16][H];
  int lane = threadIdx.x & 31, wave = threadIdx.x >> 5;
  int tile = blockIdx.x * 8 + wave;
  if (tile >= numTiles) return;
  int base = tile * 16;
  int n0 = lane & 15, hf = lane >> 4;
  half_t* M = &lds[wave][0][0];

  v16h ha[4];
#pragma unroll
  for (int kc = 0; kc < 4; ++kc)
    ha[kc] = load_afrag(hb + (size_t)base * H, H, kc * 32, lane);

  float inv[8];
#pragma unroll
  for (int r = 0; r < 8; ++r)
    inv[r] = 1.0f / fmaxf(deg[base + r + 8 * hf], 1.0f);

  // m phase
  for (int f = 0; f < 8; ++f) {
    float bb = broot[f * 16 + n0];
    v8f c;
#pragma unroll
    for (int r = 0; r < 8; ++r)
      c[r] = agg[(size_t)(base + r + 8 * hf) * H + f * 16 + n0] * inv[r] + bb;
#pragma unroll
    for (int kc = 0; kc < 4; ++kc)
      c = wmma(ha[kc], load_bfrag(wrootT, H, f * 16, kc * 32, lane), c);
#pragma unroll
    for (int r = 0; r < 8; ++r)
      M[(r + 8 * hf) * H + f * 16 + n0] = (half_t)lrelu(c[r]);
  }
  v16h ma[4];
#pragma unroll
  for (int kc = 0; kc < 4; ++kc) ma[kc] = load_afrag(M, H, kc * 32, lane);

  // gate phase
  for (int f = 0; f < 8; ++f) {
    int col = f * 16 + n0;
    v8f gr = splat8(bih[col]);
    v8f hr = splat8(bhh[col]);
    v8f gz = splat8(bih[H + col]);
    v8f hz = splat8(bhh[H + col]);
    v8f gn = splat8(bih[2 * H + col]);
    v8f hn = splat8(bhh[2 * H + col]);
#pragma unroll
    for (int kc = 0; kc < 4; ++kc) {
      gr = wmma(ma[kc], load_bfrag(wihT, H, 0 * H + f * 16, kc * 32, lane), gr);
      hr = wmma(ha[kc], load_bfrag(whhT, H, 0 * H + f * 16, kc * 32, lane), hr);
      gz = wmma(ma[kc], load_bfrag(wihT, H, 1 * H + f * 16, kc * 32, lane), gz);
      hz = wmma(ha[kc], load_bfrag(whhT, H, 1 * H + f * 16, kc * 32, lane), hz);
      gn = wmma(ma[kc], load_bfrag(wihT, H, 2 * H + f * 16, kc * 32, lane), gn);
      hn = wmma(ha[kc], load_bfrag(whhT, H, 2 * H + f * 16, kc * 32, lane), hn);
    }
#pragma unroll
    for (int r = 0; r < 8; ++r) {
      float rg = sigmoidf_(gr[r] + hr[r]);
      float zg = sigmoidf_(gz[r] + hz[r]);
      float ng = tanhf_(gn[r] + rg * hn[r]);
      size_t idx = (size_t)(base + r + 8 * hf) * H + f * 16 + n0;
      float ho = h[idx];
      float hv = (1.f - zg) * ng + zg * ho;
      h[idx] = hv;
      hb[idx] = (half_t)hv;
    }
  }
}

// ---------------- stem head ----------------------------------------------

__global__ __launch_bounds__(128) void k_stem(
    const int* __restrict__ stems_batch, const int* __restrict__ stems_local,
    const int* __restrict__ node_slices, const int* __restrict__ stem_type,
    const half_t* __restrict__ stemE, const half_t* __restrict__ hb,
    const half_t* __restrict__ w1T, const float* __restrict__ b1,
    const half_t* __restrict__ w2T, const float* __restrict__ b2,
    const half_t* __restrict__ w3T, const float* __restrict__ b3,
    float* __restrict__ out_stem, int numTiles) {
  __shared__ __align__(16) half_t lds[4][16][2 * H];  // 32 KB
  int lane = threadIdx.x & 31, wave = threadIdx.x >> 5;
  int tile = blockIdx.x * 4 + wave;
  if (tile >= numTiles) return;
  int base = tile * 16;
  int n0 = lane & 15, hf = lane >> 4;
  half_t* X = &lds[wave][0][0];

  for (int r = 0; r < 16; ++r) {
    int s = base + r;
    int node = node_slices[stems_batch[s]] + stems_local[s];
    ((uint2*)(X + r * 2 * H))[lane] = ((const uint2*)(hb + (size_t)node * H))[lane];
    ((uint2*)(X + r * 2 * H + H))[lane] =
        ((const uint2*)(stemE + (size_t)stem_type[s] * H))[lane];
  }
  v16h a1[8];
#pragma unroll
  for (int kc = 0; kc < 8; ++kc) a1[kc] = load_afrag(X, 2 * H, kc * 32, lane);
  for (int f = 0; f < 8; ++f) {
    v8f c = splat8(b1[f * 16 + n0]);
#pragma unroll
    for (int kc = 0; kc < 8; ++kc)
      c = wmma(a1[kc], load_bfrag(w1T, 2 * H, f * 16, kc * 32, lane), c);
#pragma unroll
    for (int r = 0; r < 8; ++r)
      X[(r + 8 * hf) * 2 * H + f * 16 + n0] = (half_t)lrelu(c[r]);
  }
  v16h a2[4];
#pragma unroll
  for (int kc = 0; kc < 4; ++kc) a2[kc] = load_afrag(X, 2 * H, kc * 32, lane);
  for (int f = 0; f < 8; ++f) {
    v8f c = splat8(b2[f * 16 + n0]);
#pragma unroll
    for (int kc = 0; kc < 4; ++kc)
      c = wmma(a2[kc], load_bfrag(w2T, H, f * 16, kc * 32, lane), c);
#pragma unroll
    for (int r = 0; r < 8; ++r)
      X[(r + 8 * hf) * 2 * H + f * 16 + n0] = (half_t)lrelu(c[r]);
  }
  v16h a3[4];
#pragma unroll
  for (int kc = 0; kc < 4; ++kc) a3[kc] = load_afrag(X, 2 * H, kc * 32, lane);
  for (int f = 0; f < 7; ++f) {
    int col = f * 16 + n0;
    float bb = (col < OUT_STEM) ? b3[col] : 0.f;
    v8f c = splat8(bb);
#pragma unroll
    for (int kc = 0; kc < 4; ++kc)
      c = wmma(a3[kc], load_bfrag(w3T, H, f * 16, kc * 32, lane), c);
    if (col < OUT_STEM) {
#pragma unroll
      for (int r = 0; r < 8; ++r)
        out_stem[(size_t)(base + r + 8 * hf) * OUT_STEM + col] = c[r];
    }
  }
}

// ---------------- graph mean pooling + mol head --------------------------

__global__ void k_gmean(const float* __restrict__ h, float* __restrict__ gm,
                        half_t* __restrict__ gmh) {
  int i = blockIdx.x * blockDim.x + threadIdx.x;
  if (i >= G * H) return;
  int g = i >> 7;
  int f = i & (H - 1);
  const float* p = h + (size_t)g * NPG * H + f;
  float s = 0.f;
#pragma unroll
  for (int k = 0; k < NPG; ++k) s += p[k * H];
  s *= (1.0f / NPG);
  gm[i] = s;
  gmh[i] = (half_t)s;
}

__global__ __launch_bounds__(256) void k_mol(
    const half_t* __restrict__ gmh, const half_t* __restrict__ w1T,
    const float* __restrict__ b1, const float* __restrict__ w2,
    const float* __restrict__ b2, float* __restrict__ out_mol, int numTiles) {
  int lane = threadIdx.x & 31, wave = threadIdx.x >> 5;
  int tile = blockIdx.x * 8 + wave;
  if (tile >= numTiles) return;
  int base = tile * 16;
  int n0 = lane & 15, hf = lane >> 4;
  v16h a[4];
#pragma unroll
  for (int kc = 0; kc < 4; ++kc)
    a[kc] = load_afrag(gmh + (size_t)base * H, H, kc * 32, lane);
  float acc[8];
#pragma unroll
  for (int r = 0; r < 8; ++r) acc[r] = 0.f;
  for (int f = 0; f < 8; ++f) {
    v8f c = splat8(b1[f * 16 + n0]);
#pragma unroll
    for (int kc = 0; kc < 4; ++kc)
      c = wmma(a[kc], load_bfrag(w1T, H, f * 16, kc * 32, lane), c);
    float wv = w2[f * 16 + n0];
#pragma unroll
    for (int r = 0; r < 8; ++r) acc[r] += lrelu(c[r]) * wv;
  }
#pragma unroll
  for (int r = 0; r < 8; ++r) {
    float v = acc[r];
#pragma unroll
    for (int off = 1; off < 16; off <<= 1) v += __shfl_xor(v, off, 32);
    acc[r] = v;
  }
  if (n0 == 0) {
#pragma unroll
    for (int r = 0; r < 8; ++r) out_mol[base + r + 8 * hf] = acc[r] + b2[0];
  }
}

// ---------------- host launcher ------------------------------------------

extern "C" void kernel_launch(void* const* d_in, const int* in_sizes, int n_in,
                              void* d_out, int out_size, void* d_ws, size_t ws_size,
                              hipStream_t stream) {
  (void)in_sizes; (void)n_in; (void)out_size; (void)ws_size;
  const int* node_frag_ids = (const int*)d_in[0];
  const int* edge_src      = (const int*)d_in[1];
  const int* edge_dst      = (const int*)d_in[2];
  const int* edge_bond     = (const int*)d_in[3];
  const int* stem_type     = (const int*)d_in[4];
  const int* stems_local   = (const int*)d_in[5];
  const int* stems_batch   = (const int*)d_in[6];
  const int* node_slices   = (const int*)d_in[8];
  const float* frag_emb = (const float*)d_in[9];
  const float* stem_emb = (const float*)d_in[10];
  const float* bond_emb = (const float*)d_in[11];
  const float* w_root   = (const float*)d_in[12];
  const float* b_root   = (const float*)d_in[13];
  const float* gru_wih  = (const float*)d_in[14];
  const float* gru_whh  = (const float*)d_in[15];
  const float* gru_bih  = (const float*)d_in[16];
  const float* gru_bhh  = (const float*)d_in[17];
  const float* f2e_w1   = (const float*)d_in[18];
  const float* f2e_b1   = (const float*)d_in[19];
  const float* f2e_w2   = (const float*)d_in[20];
  const float* f2e_b2   = (const float*)d_in[21];
  const float* s2p_w1   = (const float*)d_in[22];
  const float* s2p_b1   = (const float*)d_in[23];
  const float* s2p_w2   = (const float*)d_in[24];
  const float* s2p_b2   = (const float*)d_in[25];
  const float* s2p_w3   = (const float*)d_in[26];
  const float* s2p_b3   = (const float*)d_in[27];
  const float* g2p_w1   = (const float*)d_in[28];
  const float* g2p_b1   = (const float*)d_in[29];
  const float* g2p_w2   = (const float*)d_in[30];
  const float* g2p_b2   = (const float*)d_in[31];

  float* out_mol  = (float*)d_out;
  float* out_stem = (float*)d_out + G;

  char* ws = (char*)d_ws;
  size_t off = 0;
  auto alloc = [&](size_t bytes) -> void* {
    void* p = ws + off;
    off += (bytes + 255) & ~(size_t)255;
    return p;
  };
  float*  h      = (float*)alloc(sizeof(float) * (size_t)N * H);
  half_t* hbuf   = (half_t*)alloc(sizeof(half_t) * (size_t)N * H);
  float*  agg    = (float*)alloc(sizeof(float) * (size_t)N * H);
  float*  deg    = (float*)alloc(sizeof(float) * N);
  float*  gm     = (float*)alloc(sizeof(float) * (size_t)G * H);
  half_t* gmh    = (half_t*)alloc(sizeof(half_t) * (size_t)G * H);
  half_t* fragE  = (half_t*)alloc(sizeof(half_t) * 73 * H);
  half_t* stemE  = (half_t*)alloc(sizeof(half_t) * 106 * H);
  half_t* w1fT   = (half_t*)alloc(sizeof(half_t) * H * H);
  half_t* w2fT   = (half_t*)alloc(sizeof(half_t) * H * H);
  half_t* wrootT = (half_t*)alloc(sizeof(half_t) * H * H);
  half_t* wihT   = (half_t*)alloc(sizeof(half_t) * 3 * H * H);
  half_t* whhT   = (half_t*)alloc(sizeof(half_t) * 3 * H * H);
  half_t* s1T    = (half_t*)alloc(sizeof(half_t) * H * 2 * H);
  half_t* s2T    = (half_t*)alloc(sizeof(half_t) * H * H);
  half_t* w3T    = (half_t*)alloc(sizeof(half_t) * 112 * H);
  half_t* g1T    = (half_t*)alloc(sizeof(half_t) * H * H);

  auto cdiv = [](int a, int b) { return (a + b - 1) / b; };

  // degrees (constant across steps)
  k_zero_f32<<<cdiv(N, 256), 256, 0, stream>>>(deg, N);
  k_deg<<<cdiv(E, 256), 256, 0, stream>>>(edge_dst, deg, E);

  // one-time f16 conversion; B-operands packed [N][K]
  k_cvt<<<cdiv(73 * H, 256), 256, 0, stream>>>(frag_emb, fragE, 73 * H);
  k_cvt<<<cdiv(106 * H, 256), 256, 0, stream>>>(stem_emb, stemE, 106 * H);
  k_cvt<<<cdiv(3 * H * H, 256), 256, 0, stream>>>(gru_wih, wihT, 3 * H * H);
  k_cvt<<<cdiv(3 * H * H, 256), 256, 0, stream>>>(gru_whh, whhT, 3 * H * H);
  k_cvtT<<<cdiv(H * H, 256), 256, 0, stream>>>(f2e_w1, w1fT, H, H);
  k_cvtT<<<cdiv(H * H, 256), 256, 0, stream>>>(f2e_w2, w2fT, H, H);
  k_cvtT<<<cdiv(H * H, 256), 256, 0, stream>>>(w_root, wrootT, H, H);
  k_cvtT<<<cdiv(2 * H * H, 256), 256, 0, stream>>>(s2p_w1, s1T, 2 * H, H);
  k_cvtT<<<cdiv(H * H, 256), 256, 0, stream>>>(s2p_w2, s2T, H, H);
  k_cvt_w3<<<cdiv(112 * H, 256), 256, 0, stream>>>(s2p_w3, w3T);
  k_cvtT<<<cdiv(H * H, 256), 256, 0, stream>>>(g2p_w1, g1T, H, H);

  const int nodeTiles = N / 16;  // 12500, exact
  k_embed_f2e<<<cdiv(nodeTiles, 8), 256, 0, stream>>>(
      node_frag_ids, fragE, w1fT, f2e_b1, w2fT, f2e_b2, h, hbuf, nodeTiles);

  for (int step = 0; step < STEPS; ++step) {
    k_edge_agg<<<G, 256, 0, stream>>>(edge_src, edge_dst, edge_bond, bond_emb,
                                      h, agg);
    k_gru<<<cdiv(nodeTiles, 8), 256, 0, stream>>>(
        agg, deg, wrootT, b_root, wihT, gru_bih, whhT, gru_bhh, h, hbuf,
        nodeTiles);
  }

  const int stemTiles = S / 16;  // 6250, exact
  k_stem<<<cdiv(stemTiles, 4), 128, 0, stream>>>(
      stems_batch, stems_local, node_slices, stem_type, stemE, hbuf, s1T,
      s2p_b1, s2T, s2p_b2, w3T, s2p_b3, out_stem, stemTiles);

  k_gmean<<<cdiv(G * H, 256), 256, 0, stream>>>(h, gm, gmh);
  const int molTiles = G / 16;  // 625, exact
  k_mol<<<cdiv(molTiles, 8), 256, 0, stream>>>(gmh, g1T, g2p_b1, g2p_w2,
                                               g2p_b2, out_mol, molTiles);
}
